// DenseNibblePPR_60224031424549
// MI455X (gfx1250) — matrix operand order repaired and checked by reading, not compiled, not run on hardware
//
#include <hip/hip_runtime.h>
#include <hip/hip_bf16.h>

// Problem dimensions (fixed by the reference)
#define N_NODES 16384   // N
#define D_IN    128
#define D_H     64
#define B_SEEDS 4096

typedef __attribute__((ext_vector_type(16))) __bf16        v16bf;
typedef __attribute__((ext_vector_type(2)))  __bf16        v2bf;
typedef __attribute__((ext_vector_type(8)))  float         v8f;
typedef __attribute__((ext_vector_type(4)))  float         v4f;
typedef __attribute__((ext_vector_type(4)))  unsigned int  v4u;

union Frag16 {
    v16bf        bf;
    unsigned int u[8];
    v4u          q[2];
};

// f32 pair -> packed bf16 (lo in [15:0], hi in [31:16]).
// Preference: hardware packed converter; fallback: single v_perm_b32
// byte-select of the two high halves (truncation to bf16).
__device__ __forceinline__ unsigned int pk_bf16(float lo, float hi) {
#if __has_builtin(__builtin_amdgcn_cvt_pk_bf16_f32)
    v2bf r = __builtin_amdgcn_cvt_pk_bf16_f32(lo, hi);
    return __builtin_bit_cast(unsigned int, r);
#else
    // result bytes [3:2] = hi[31:16], [1:0] = lo[31:16]
    return __builtin_amdgcn_perm(__float_as_uint(hi), __float_as_uint(lo),
                                 0x07060302u);
#endif
}

// ---------------------------------------------------------------------------
// Kernel 1: encT[f][k] = (X @ W + b)[k][f] in bf16.
// One wave per 16-row (node) tile of X; K = 128 (4 WMMA steps); N = 64 (4 tiles).
// ---------------------------------------------------------------------------
__global__ void __launch_bounds__(256)
encode_kernel(const float* __restrict__ X,
              const float* __restrict__ W,
              const float* __restrict__ bias,
              unsigned short* __restrict__ encT) {
    const int lane  = threadIdx.x & 31;
    const int wave  = threadIdx.x >> 5;
    const int mtile = blockIdx.x * 8 + wave;       // 1024 tiles of 16 nodes
    const int half  = lane >> 4;                   // lane half selects K sub-ranges
    const int l16   = lane & 15;

    const float* xrow = X + (size_t)(mtile * 16 + l16) * D_IN;

    v8f acc[4] = {};

    for (int kb = 0; kb < D_IN; kb += 32) {
        // A fragment (16x32 bf16): lane half h holds K = kb+16h..+15
        Frag16 a;
        {
            const v4f* p0 = (const v4f*)(xrow + kb + half * 16);
            const v4f* p1 = (const v4f*)(xrow + kb + half * 16 + 8);
            v4f f0 = p0[0], f1 = p0[1];
            v4f g0 = p1[0], g1 = p1[1];
            a.u[0] = pk_bf16(f0[0], f0[1]); a.u[1] = pk_bf16(f0[2], f0[3]);
            a.u[2] = pk_bf16(f1[0], f1[1]); a.u[3] = pk_bf16(f1[2], f1[3]);
            a.u[4] = pk_bf16(g0[0], g0[1]); a.u[5] = pk_bf16(g0[2], g0[3]);
            a.u[6] = pk_bf16(g1[0], g1[1]); a.u[7] = pk_bf16(g1[2], g1[3]);
        }
#pragma unroll
        for (int t = 0; t < 4; ++t) {
            // B fragment (32x16 bf16): lane holds col n, K = kb + 16*half + 0..15
            Frag16 bf;
            const float* wcol = W + (size_t)(kb + half * 16) * D_H + t * 16 + l16;
#pragma unroll
            for (int v = 0; v < 8; ++v)
                bf.u[v] = pk_bf16(wcol[(2 * v) * D_H], wcol[(2 * v + 1) * D_H]);
            acc[t] = __builtin_amdgcn_wmma_f32_16x16x32_bf16(
                false, a.bf, false, bf.bf, (short)0, acc[t], false, false);
        }
    }

    // C layout: acc[t][v] = enc(node = mtile*16 + v + 8*half, f = t*16 + l16).
    // 8 consecutive nodes per lane -> pack to one 128-bit bf16 store into encT[f][...]
#pragma unroll
    for (int t = 0; t < 4; ++t) {
        const int   f  = t * 16 + l16;
        const float bv = bias[f];
        v4u pk;
        pk[0] = pk_bf16(acc[t][0] + bv, acc[t][1] + bv);
        pk[1] = pk_bf16(acc[t][2] + bv, acc[t][3] + bv);
        pk[2] = pk_bf16(acc[t][4] + bv, acc[t][5] + bv);
        pk[3] = pk_bf16(acc[t][6] + bv, acc[t][7] + bv);
        *(v4u*)(encT + (size_t)f * N_NODES + mtile * 16 + half * 8) = pk;
    }
}

// ---------------------------------------------------------------------------
// Kernel 2: out[m][f] = sum_k ppr[idx[m]][k] * encT[f][k]
// One workgroup (8 waves) per 16-row output tile; waves split K 8 ways,
// LDS reduction of the 16x64 f32 partials. ppr streamed nontemporally.
// ---------------------------------------------------------------------------
__global__ void __launch_bounds__(256)
diffuse_kernel(const float* __restrict__ ppr,
               const unsigned short* __restrict__ encT,
               const int* __restrict__ idx,
               float* __restrict__ out) {
    __shared__ float red[8 * 1024];                // 32 KB

    const int lane  = threadIdx.x & 31;
    const int wave  = threadIdx.x >> 5;            // split-K index 0..7
    const int mtile = blockIdx.x;                  // 0..255
    const int half  = lane >> 4;
    const int l16   = lane & 15;

    const int row = idx[mtile * 16 + l16];         // gathered seed node
    const float* prow = ppr + (size_t)row * N_NODES;

    v8f acc[4] = {};

    const int k0 = wave * (N_NODES / 8);
    const int k1 = k0 + (N_NODES / 8);
#pragma unroll 2
    for (int kb = k0; kb < k1; kb += 32) {
        // A fragment from ppr row (read-once stream -> nontemporal b128 loads)
        Frag16 a;
        {
            const v4f* p0 = (const v4f*)(prow + kb + half * 16);
            const v4f* p1 = (const v4f*)(prow + kb + half * 16 + 8);
            v4f f0 = __builtin_nontemporal_load(p0);
            v4f f1 = __builtin_nontemporal_load(p0 + 1);
            v4f g0 = __builtin_nontemporal_load(p1);
            v4f g1 = __builtin_nontemporal_load(p1 + 1);
            a.u[0] = pk_bf16(f0[0], f0[1]); a.u[1] = pk_bf16(f0[2], f0[3]);
            a.u[2] = pk_bf16(f1[0], f1[1]); a.u[3] = pk_bf16(f1[2], f1[3]);
            a.u[4] = pk_bf16(g0[0], g0[1]); a.u[5] = pk_bf16(g0[2], g0[3]);
            a.u[6] = pk_bf16(g1[0], g1[1]); a.u[7] = pk_bf16(g1[2], g1[3]);
        }
        // B fragments from pre-converted bf16 encT (L2-resident, contiguous)
        Frag16 bf[4];
#pragma unroll
        for (int t = 0; t < 4; ++t) {
            const v4u* bp = (const v4u*)(encT +
                (size_t)(t * 16 + l16) * N_NODES + kb + half * 16);
            bf[t].q[0] = bp[0];
            bf[t].q[1] = bp[1];
        }
#pragma unroll
        for (int t = 0; t < 4; ++t)
            acc[t] = __builtin_amdgcn_wmma_f32_16x16x32_bf16(
                false, a.bf, false, bf[t].bf, (short)0, acc[t], false, false);
    }

    // Dump per-wave partial C tiles to LDS: red[wave][slot = t*8+v][lane]
#pragma unroll
    for (int t = 0; t < 4; ++t)
#pragma unroll
        for (int v = 0; v < 8; ++v)
            red[wave * 1024 + (t * 8 + v) * 32 + lane] = acc[t][v];
    __syncthreads();

    // 256 threads reduce 8 partials each over 4 slots and store f32 output.
    const int rlane = threadIdx.x & 31;
    const int s0    = (threadIdx.x >> 5) * 4;
#pragma unroll
    for (int j = 0; j < 4; ++j) {
        const int s = s0 + j;                      // slot 0..31
        float sum = 0.0f;
#pragma unroll
        for (int w = 0; w < 8; ++w)
            sum += red[w * 1024 + s * 32 + rlane];
        const int t = s >> 3, v = s & 7;
        const int m = mtile * 16 + v + 8 * (rlane >> 4);
        const int n = t * 16 + (rlane & 15);
        out[(size_t)m * D_H + n] = sum;
    }
}

// ---------------------------------------------------------------------------
extern "C" void kernel_launch(void* const* d_in, const int* in_sizes, int n_in,
                              void* d_out, int out_size, void* d_ws, size_t ws_size,
                              hipStream_t stream) {
    const float* X    = (const float*)d_in[0];
    const float* ppr  = (const float*)d_in[1];
    const float* W    = (const float*)d_in[2];
    const float* bias = (const float*)d_in[3];
    const int*   idx  = (const int*)d_in[4];
    float*       out  = (float*)d_out;

    unsigned short* encT = (unsigned short*)d_ws;  // 64 * 16384 bf16 = 2 MB

    // enc = X @ W + b, stored transposed bf16
    encode_kernel<<<(N_NODES / 16) / 8, 256, 0, stream>>>(X, W, bias, encT);
    // out = ppr[idx] @ enc
    diffuse_kernel<<<B_SEEDS / 16, 256, 0, stream>>>(ppr, encT, idx, out);
}
